// BiLSTM_CRF_9208409882830
// MI455X (gfx1250) — compile-verified, matrix operand order checked by reading, not run on hardware
//
#include <hip/hip_runtime.h>
#include <hip/hip_bf16.h>
#include <stdint.h>

// ---------------------------------------------------------------------------
// BiLSTM-CRF inference for MI455X (gfx1250, wave32, WMMA 16x16x32 f16).
//
//   1) cvt_f16:      f32 -> f16 weight conversion
//   2) embed_gather: embedding lookup -> f16 activation matrix [T*B, E]
//   3) input_gemm:   gatesX[dir] = emb @ W_ih^T + b   (async-LDS pipelined WMMA)
//   4) lstm_recur:   sequential recurrence, h/c/gates in LDS, WMMA per step
//   5) emit_gemm:    emissions = h @ W_lin^T + b_lin  (WMMA, N=16)
//   6) viterbi:      one wave per batch row, shuffle-based max/argmax
// ---------------------------------------------------------------------------

#define B_   32
#define T_   256
#define E_   512
#define HD_  512
#define H_   1024
#define G4_  2048            // 4*HD
#define K_   16
#define MROWS (T_*B_)        // 8192

typedef __attribute__((ext_vector_type(16))) _Float16 v16h;
typedef __attribute__((ext_vector_type(8)))  _Float16 v8h;
typedef __attribute__((ext_vector_type(8)))  float    v8f;

// Load one f16 WMMA fragment (A row / B^T column) for this lane.
// K is contiguous in memory. Per ISA layout: lane<16 holds K = k0+0..7 and
// k0+16..23; lane>=16 holds K = k0+8..15 and k0+24..31 (hf = lane>>4).
template <typename PtrT>
__device__ __forceinline__ v16h ld_frag(PtrT row_ptr, int k0, int hf) {
  union { v16h v; v8h h[2]; } u;
  const _Float16* p = (const _Float16*)row_ptr + k0 + hf * 8;
  u.h[0] = *(const v8h*)(p);
  u.h[1] = *(const v8h*)(p + 16);
  return u.v;
}

__device__ __forceinline__ float sigf(float x) {
  return 1.0f / (1.0f + __expf(-x));
}
__device__ __forceinline__ float tanh_fast(float x) {
  return 1.0f - 2.0f / (__expf(2.0f * x) + 1.0f);
}

// --- CDNA5 async global->LDS DMA (ASYNCcnt-tracked), inline asm -------------
// VDST = wave-relative LDS byte address (low 32 bits of the shared-aperture
// generic address, per ISA 10.2).
__device__ __forceinline__ void async_copy_b128(void* lds_dst, const void* gsrc) {
  const unsigned lds_off = (unsigned)(size_t)lds_dst;
  asm volatile("global_load_async_to_lds_b128 %0, %1, off"
               :: "v"(lds_off), "v"(gsrc)
               : "memory");
}
__device__ __forceinline__ void wait_asynccnt(int n) {
  asm volatile("s_wait_asynccnt %0" :: "i"(n) : "memory");
}

// ---------------------------------------------------------------- 1) convert
__global__ void cvt_f16(const float* __restrict__ src,
                        _Float16* __restrict__ dst, int n) {
  for (int i = blockIdx.x * blockDim.x + threadIdx.x; i < n;
       i += gridDim.x * blockDim.x)
    dst[i] = (_Float16)src[i];
}

// ------------------------------------------------------- 2) embedding gather
__global__ void embed_gather(const int* __restrict__ x,
                             const float* __restrict__ embed,
                             _Float16* __restrict__ emb) {
  const size_t total = (size_t)MROWS * E_;
  for (size_t i = (size_t)blockIdx.x * blockDim.x + threadIdx.x; i < total;
       i += (size_t)gridDim.x * blockDim.x) {
    size_t row = i >> 9;              // / E_
    int    e   = (int)(i & (E_ - 1));
    int    t   = (int)(row >> 5);     // / B_
    int    b   = (int)(row & (B_ - 1));
    int    tok = x[b * T_ + t];
    emb[i] = (_Float16)embed[(size_t)tok * E_ + e];
  }
}

// ------------------------------------------------------- 3) input-side GEMM
// gatesX[dir][m][n] = sum_k emb[m][k] * W_ih[dir][n][k] + bias[n]  (f16 out)
// Block 256 thr = 8 waves; tile 64(M) x 128(N); K pipelined through LDS with
// double-buffered global_load_async_to_lds_b128 (ASYNCcnt).
#define KCH   (E_ / 32)      // 16 K-chunks
#define LROW  40             // padded LDS row stride in halves (80B: bank-safe)
__global__ __launch_bounds__(256)
void input_gemm(const _Float16* __restrict__ A,        // [8192][512]
                const _Float16* __restrict__ W0,       // [2048][512] f16
                const _Float16* __restrict__ W1,
                const float* __restrict__ bias0,
                const float* __restrict__ bias1,
                _Float16* __restrict__ out0,           // [8192][2048] f16
                _Float16* __restrict__ out1) {
  __shared__ _Float16 Ab[2][64 * LROW];    // 2 x 5 KB
  __shared__ _Float16 Bb[2][128 * LROW];   // 2 x 10 KB

  const int dir  = blockIdx.z;
  const _Float16* W    = dir ? W1 : W0;
  const float*    bias = dir ? bias1 : bias0;
  _Float16*       out  = dir ? out1 : out0;

  const int tid  = threadIdx.x;
  const int wave = tid >> 5;
  const int lane = tid & 31;
  const int r    = lane & 15;
  const int hf   = lane >> 4;

  const int gm0 = blockIdx.y * 64;
  const int gn0 = blockIdx.x * 128;
  const int m0l = (wave & 3) * 16;         // local M tile
  const int n0l = (wave >> 2) * 64;        // local N strip (4 tiles)

  // stage one 32-wide K chunk of the A (64 rows) and B (128 rows) tiles
  auto issue_chunk = [&](int kc, int buf) {
    const int k0 = kc * 32;
    {   // A tile: 64 rows x 64B = 256 x b128, one per thread
      const int row = tid >> 2, seg = tid & 3;
      async_copy_b128(&Ab[buf][row * LROW + seg * 8],
                      A + (size_t)(gm0 + row) * E_ + k0 + seg * 8);
    }
#pragma unroll
    for (int i = 0; i < 2; ++i) {   // B tile: 128 rows x 64B = 512 x b128
      const int id = tid + i * 256;
      const int row = id >> 2, seg = id & 3;
      async_copy_b128(&Bb[buf][row * LROW + seg * 8],
                      W + (size_t)(gn0 + row) * E_ + k0 + seg * 8);
    }
  };

  issue_chunk(0, 0);

  v8f acc[4] = {};
  for (int kc = 0; kc < KCH; ++kc) {
    const int cur = kc & 1;
    if (kc + 1 < KCH) {
      issue_chunk(kc + 1, cur ^ 1);
      wait_asynccnt(3);              // 3 outstanding = next chunk only
    } else {
      wait_asynccnt(0);
    }
    __syncthreads();                 // all waves' DMA for `cur` complete

    const _Float16* arow = &Ab[cur][(m0l + r) * LROW];
    v16h a = ld_frag(arow, 0, hf);   // ds_load_b128 x2
#pragma unroll
    for (int j = 0; j < 4; ++j) {
      const _Float16* brow = &Bb[cur][(n0l + j * 16 + r) * LROW];
      v16h b = ld_frag(brow, 0, hf);
      acc[j] = __builtin_amdgcn_wmma_f32_16x16x32_f16(
          false, a, false, b, (short)0, acc[j], false, false);
    }
    __syncthreads();                 // protect buffer reuse next iteration
  }

#pragma unroll
  for (int j = 0; j < 4; ++j) {
    const int n  = gn0 + n0l + j * 16 + r;
    const float bv = bias[n];
#pragma unroll
    for (int q = 0; q < 8; ++q) {
      const int m = gm0 + m0l + q + hf * 8;
      out[(size_t)m * G4_ + n] = (_Float16)(acc[j][q] + bv);
    }
  }
}

// ---------------------------------------------------- 4) recurrent LSTM core
// One persistent workgroup per direction. h (f16) + c (f32) + gate buffer
// (f16) live in LDS (32+64+128 = 224 KB of the 320 KB WGP LDS).
__global__ __launch_bounds__(512)
void lstm_recur(const _Float16* __restrict__ Whh0,     // [2048][512] f16
                const _Float16* __restrict__ Whh1,
                const _Float16* __restrict__ gX0,      // [T][B][2048] f16
                const _Float16* __restrict__ gX1,
                _Float16* __restrict__ h_out) {        // [T][B][1024] f16
  extern __shared__ char smem[];
  _Float16* h_sh = (_Float16*)smem;                        // 32*512  f16
  _Float16* g_sh = (_Float16*)(smem + B_ * HD_ * 2);       // 32*2048 f16
  float*    c_sh = (float*)(smem + B_ * HD_ * 2 + B_ * G4_ * 2);

  const int dir = blockIdx.x;
  const _Float16* W  = dir ? Whh1 : Whh0;
  const _Float16* gX = dir ? gX1 : gX0;

  const int tid  = threadIdx.x;           // 512 threads = 16 waves
  const int wave = tid >> 5;
  const int lane = tid & 31;
  const int r    = lane & 15;
  const int hf   = lane >> 4;

  for (int e = tid; e < B_ * HD_; e += 512) {
    h_sh[e] = (_Float16)0.0f;
    c_sh[e] = 0.0f;
  }
  __syncthreads();

  for (int step = 0; step < T_; ++step) {
    const int t = dir ? (T_ - 1 - step) : step;
    const _Float16* gXt = gX + (size_t)t * B_ * G4_;

    // --- WMMA phase: g = gatesX[t] + h @ W_hh^T -------------------------
    for (int i = 0; i < 16; ++i) {
      const int idx = i * 16 + wave;          // 0..255
      const int m0  = (idx >> 7) * 16;        // 0 or 16
      const int n0  = (idx & 127) * 16;
      const int n   = n0 + r;

      v8f acc;
#pragma unroll
      for (int q = 0; q < 8; ++q)
        acc[q] = (float)gXt[(size_t)(m0 + q + hf * 8) * G4_ + n];

      const _Float16* hrow = h_sh + (size_t)(m0 + r) * HD_;
      const _Float16* wrow = W + (size_t)n * HD_;
      __builtin_prefetch(wrow, 0, 3);

      // 2-stage software pipeline over K so loads overlap WMMA
      v16h a = ld_frag(hrow, 0, hf);          // ds_load_b128 x2
      v16h b = ld_frag(wrow, 0, hf);          // global (L2-resident W_hh)
#pragma unroll
      for (int k0 = 0; k0 < HD_; k0 += 32) {
        v16h an = a, bn = b;
        if (k0 + 32 < HD_) {
          an = ld_frag(hrow, k0 + 32, hf);
          bn = ld_frag(wrow, k0 + 32, hf);
        }
        acc = __builtin_amdgcn_wmma_f32_16x16x32_f16(
            false, a, false, b, (short)0, acc, false, false);
        a = an; b = bn;
      }
#pragma unroll
      for (int q = 0; q < 8; ++q)
        g_sh[(size_t)(m0 + q + hf * 8) * G4_ + n] = (_Float16)acc[q];
    }
    __syncthreads();

    // --- pointwise phase: i,f,g,o -> c,h --------------------------------
    for (int e = tid; e < B_ * HD_; e += 512) {
      const int b  = e >> 9;
      const int hd = e & (HD_ - 1);
      const float gi = (float)g_sh[b * G4_ + hd];
      const float gf = (float)g_sh[b * G4_ + HD_ + hd];
      const float gg = (float)g_sh[b * G4_ + 2 * HD_ + hd];
      const float go = (float)g_sh[b * G4_ + 3 * HD_ + hd];
      const float c  = sigf(gf) * c_sh[e] + sigf(gi) * tanh_fast(gg);
      const float h  = sigf(go) * tanh_fast(c);
      c_sh[e] = c;
      h_sh[e] = (_Float16)h;
      h_out[((size_t)t * B_ + b) * H_ + dir * HD_ + hd] = (_Float16)h;
    }
    __syncthreads();
  }
}

// ------------------------------------------------------- 5) emission GEMM
__global__ __launch_bounds__(256)
void emit_gemm(const _Float16* __restrict__ Hm,        // [8192][1024]
               const _Float16* __restrict__ Wlin,      // [16][1024] f16
               const float* __restrict__ b_lin,
               float* __restrict__ em) {               // [8192][16] f32
  const int wave = threadIdx.x >> 5;
  const int lane = threadIdx.x & 31;
  const int r    = lane & 15;
  const int hf   = lane >> 4;
  const int m0   = blockIdx.x * 128 + wave * 16;

  const _Float16* arow = Hm + (size_t)(m0 + r) * H_;
  const _Float16* brow = Wlin + (size_t)r * H_;

  v8f acc = {};
  v16h a = ld_frag(arow, 0, hf);
  v16h b = ld_frag(brow, 0, hf);
#pragma unroll
  for (int k0 = 0; k0 < H_; k0 += 32) {
    v16h an = a, bn = b;
    if (k0 + 32 < H_) {
      an = ld_frag(arow, k0 + 32, hf);
      bn = ld_frag(brow, k0 + 32, hf);
    }
    acc = __builtin_amdgcn_wmma_f32_16x16x32_f16(
        false, a, false, b, (short)0, acc, false, false);
    a = an; b = bn;
  }
  const float bv = b_lin[r];
#pragma unroll
  for (int q = 0; q < 8; ++q)
    em[(size_t)(m0 + q + hf * 8) * K_ + r] = acc[q] + bv;
}

// --------------------------------------------------------------- 6) viterbi
__global__ __launch_bounds__(1024)
void viterbi(const float* __restrict__ em,             // [T*B][16]
             const uint8_t* __restrict__ mask,         // [B][T]
             const float* __restrict__ start_trans,
             const float* __restrict__ end_trans,
             const float* __restrict__ trans,          // [16][16]
             int* __restrict__ hist,                   // [(T-1)*B][16]
             float* __restrict__ out) {                // tags[B*T], score[B]
  const int b    = threadIdx.x >> 5;
  const int lane = threadIdx.x & 31;
  const int j    = lane & 15;

  float score = start_trans[j] + em[(size_t)b * K_ + j];

  for (int t = 1; t < T_; ++t) {
    const float e = em[((size_t)t * B_ + b) * K_ + j];
    float best = -3.0e38f;
    int   barg = 0;
#pragma unroll
    for (int i = 0; i < K_; ++i) {
      const float si = __shfl(score, i, 32);
      const float v  = si + trans[i * K_ + j] + e;
      if (v > best) { best = v; barg = i; }
    }
    if (lane < 16)
      hist[((size_t)(t - 1) * B_ + b) * K_ + j] = barg;
    const bool m = mask[b * T_ + t] != 0;
    score = m ? best : score;
  }
  score += end_trans[j];

  float bs = -3.0e38f;
  int   bt = 0;
#pragma unroll
  for (int i = 0; i < K_; ++i) {
    const float si = __shfl(score, i, 32);
    if (si > bs) { bs = si; bt = i; }
  }

  if (lane == 0) {
    out[B_ * T_ + b] = bs;
    int cur = bt;
    out[b * T_ + (T_ - 1)] = mask[b * T_ + (T_ - 1)] ? (float)cur : 0.0f;
    for (int t = T_ - 2; t >= 0; --t) {
      const int prev = hist[((size_t)t * B_ + b) * K_ + cur];
      if (mask[b * T_ + t + 1]) cur = prev;
      out[b * T_ + t] = mask[b * T_ + t] ? (float)cur : 0.0f;
    }
  }
}

// ---------------------------------------------------------------------------
extern "C" void kernel_launch(void* const* d_in, const int* in_sizes, int n_in,
                              void* d_out, int out_size, void* d_ws,
                              size_t ws_size, hipStream_t stream) {
  const int*     x      = (const int*)d_in[0];
  const uint8_t* mask   = (const uint8_t*)d_in[1];
  const float*   embed  = (const float*)d_in[2];
  const float*   W_ih_f = (const float*)d_in[3];
  const float*   W_hh_f = (const float*)d_in[4];
  const float*   b_f    = (const float*)d_in[5];
  const float*   W_ih_b = (const float*)d_in[6];
  const float*   W_hh_b = (const float*)d_in[7];
  const float*   b_b    = (const float*)d_in[8];
  const float*   W_lin  = (const float*)d_in[9];
  const float*   b_lin  = (const float*)d_in[10];
  const float*   s_tr   = (const float*)d_in[11];
  const float*   e_tr   = (const float*)d_in[12];
  const float*   trans  = (const float*)d_in[13];
  float*         out    = (float*)d_out;

  char* ws = (char*)d_ws;
  size_t off = 0;
  auto alloc = [&](size_t bytes) {
    char* p = ws + off;
    off += (bytes + 255) & ~(size_t)255;
    return p;
  };
  _Float16* embA   = (_Float16*)alloc((size_t)MROWS * E_ * 2);
  _Float16* Wih0   = (_Float16*)alloc((size_t)G4_ * E_ * 2);
  _Float16* Wih1   = (_Float16*)alloc((size_t)G4_ * E_ * 2);
  _Float16* Whh0   = (_Float16*)alloc((size_t)G4_ * HD_ * 2);
  _Float16* Whh1   = (_Float16*)alloc((size_t)G4_ * HD_ * 2);
  _Float16* WlinH  = (_Float16*)alloc((size_t)K_ * H_ * 2);
  _Float16* gX0    = (_Float16*)alloc((size_t)MROWS * G4_ * 2);
  _Float16* gX1    = (_Float16*)alloc((size_t)MROWS * G4_ * 2);
  _Float16* h_out  = (_Float16*)alloc((size_t)MROWS * H_ * 2);
  float*    em     = (float*)alloc((size_t)MROWS * K_ * 4);
  int*      hist   = (int*)alloc((size_t)(T_ - 1) * B_ * K_ * 4);

  cvt_f16<<<512, 256, 0, stream>>>(W_ih_f, Wih0, G4_ * E_);
  cvt_f16<<<512, 256, 0, stream>>>(W_ih_b, Wih1, G4_ * E_);
  cvt_f16<<<512, 256, 0, stream>>>(W_hh_f, Whh0, G4_ * HD_);
  cvt_f16<<<512, 256, 0, stream>>>(W_hh_b, Whh1, G4_ * HD_);
  cvt_f16<<<64, 256, 0, stream>>>(W_lin, WlinH, K_ * H_);

  embed_gather<<<2048, 256, 0, stream>>>(x, embed, embA);

  input_gemm<<<dim3(G4_ / 128, MROWS / 64, 2), 256, 0, stream>>>(
      embA, Wih0, Wih1, b_f, b_b, gX0, gX1);

  const size_t lds_bytes = (size_t)B_ * HD_ * 2 +   // h  f16
                           (size_t)B_ * G4_ * 2 +   // g  f16
                           (size_t)B_ * HD_ * 4;    // c  f32
  lstm_recur<<<2, 512, lds_bytes, stream>>>(Whh0, Whh1, gX0, gX1, h_out);

  emit_gemm<<<MROWS / 128, 256, 0, stream>>>(h_out, WlinH, b_lin, em);

  viterbi<<<1, 1024, 0, stream>>>(em, mask, s_tr, e_tr, trans, hist, out);
}
